// FLASH_80900003988039
// MI455X (gfx1250) — matrix operand order repaired and checked by reading, not compiled, not run on hardware
//
#include <hip/hip_runtime.h>

typedef __attribute__((ext_vector_type(16))) _Float16 v16h;
typedef __attribute__((ext_vector_type(8)))  _Float16 v8h;
typedef __attribute__((ext_vector_type(8)))  float    v8f;
typedef __attribute__((ext_vector_type(4)))  int      v4i;

#define AS1 __attribute__((address_space(1)))
#define AS3 __attribute__((address_space(3)))

#define DIMC  256   // model dim C
#define HIDD  512   // hidden
#define SDIM  128   // S
#define NTOK  128   // tokens per window
#define UVCOLS 1152 // 2*HID + S
#define SCALE_Q 4.8828125e-4f  // 1/sqrt(256)/128

// workspace layout (halves): 8 uv K-tiles of [1152][32], then 16 proj K-tiles of [256][32]
#define WS_UV_TILE   (UVCOLS * 32)          // 36864 halves per uv K-tile
#define WS_PJ_BASE   (8 * WS_UV_TILE)       // 294912
#define WS_PJ_TILE   (DIMC * 32)            // 8192 halves per proj K-tile
#define WS_TOTAL     (WS_PJ_BASE + 16 * WS_PJ_TILE)  // 425984 halves

#if defined(__AMDGCN__) && __has_builtin(__builtin_amdgcn_global_load_async_to_lds_b128)
#define ASYNC_LDS 1
#else
#define ASYNC_LDS 0
#endif

// ---- fragment loaders (16-bit data, wave32 WMMA layouts) -------------------
__device__ __forceinline__ v16h ldA16(const _Float16* s, int rs, int mb, int k0, int lane) {
  int m  = mb + (lane & 15);
  int hi = lane >> 4;
  const _Float16* p = s + m * rs + k0 + hi * 8;
  v8h lo = *(const v8h*)(p);
  v8h hh = *(const v8h*)(p + 16);
  v16h r;
#pragma unroll
  for (int i = 0; i < 8; ++i) { r[i] = lo[i]; r[i + 8] = hh[i]; }
  return r;
}

__device__ __forceinline__ v16h ldB16(const _Float16* s, int rs, int nb, int k0, int lane) {
  int n  = nb + (lane & 15);
  int hi = lane >> 4;
  const _Float16* p = s + n * rs + k0 + hi * 8;
  v8h lo = *(const v8h*)(p);
  v8h hh = *(const v8h*)(p + 16);
  v16h r;
#pragma unroll
  for (int i = 0; i < 8; ++i) { r[i] = lo[i]; r[i + 8] = hh[i]; }
  return r;
}

__device__ __forceinline__ v8f wmma16(v16h a, v16h b, v8f c) {
  return __builtin_amdgcn_wmma_f32_16x16x32_f16(false, a, false, b, (short)0, c, false, false);
}

template <int NT>
__device__ __forceinline__ void mm_tiles(v8f* acc, v16h a, const _Float16* B,
                                         int rsB, int kB, int lane) {
#pragma unroll
  for (int j = 0; j < NT; j += 4) {
    v16h b0 = ldB16(B, rsB, (j + 0) * 16, kB, lane);
    v16h b1 = ldB16(B, rsB, (j + 1) * 16, kB, lane);
    v16h b2 = ldB16(B, rsB, (j + 2) * 16, kB, lane);
    v16h b3 = ldB16(B, rsB, (j + 3) * 16, kB, lane);
    acc[j + 0] = wmma16(a, b0, acc[j + 0]);
    acc[j + 1] = wmma16(a, b1, acc[j + 1]);
    acc[j + 2] = wmma16(a, b2, acc[j + 2]);
    acc[j + 3] = wmma16(a, b3, acc[j + 3]);
  }
}

// ---- async contiguous stage: NCH 16-byte chunks global(f16) -> LDS ---------
template <int NCH>
__device__ __forceinline__ void stage_cp(_Float16* dst, const _Float16* src, int tid) {
#pragma unroll
  for (int e = tid; e < NCH; e += 256) {
#if ASYNC_LDS
    __builtin_amdgcn_global_load_async_to_lds_b128(
        (AS1 v4i*)(uintptr_t)(src + e * 8),
        (AS3 v4i*)(uintptr_t)(dst + e * 8), 0, 0);
#else
    *(v8h*)(dst + e * 8) = *(const v8h*)(src + e * 8);
#endif
  }
}

__device__ __forceinline__ void stage_fence() {
#if ASYNC_LDS
#if __has_builtin(__builtin_amdgcn_s_wait_asynccnt)
  __builtin_amdgcn_s_wait_asynccnt(0);
#else
  asm volatile("s_wait_asynccnt 0x0" ::: "memory");
#endif
#endif
}

// ---- prep: fp32 weights -> f16, transposed (n-major), tile-contiguous ------
__global__ __launch_bounds__(256)
void prep_weights_kernel(const float* __restrict__ uv_w,
                         const float* __restrict__ proj_w,
                         _Float16* __restrict__ ws) {
  int i = blockIdx.x * 256 + threadIdx.x;
  if (i < WS_PJ_BASE) {
    int kt = i / WS_UV_TILE;
    int r  = i - kt * WS_UV_TILE;
    int n  = r >> 5, kk = r & 31;
    ws[i] = (_Float16)uv_w[(size_t)(kt * 32 + kk) * UVCOLS + n];
  } else if (i < WS_TOTAL) {
    int j  = i - WS_PJ_BASE;
    int kt = j / WS_PJ_TILE;
    int r  = j - kt * WS_PJ_TILE;
    int n  = r >> 5, kk = r & 31;
    ws[i] = (_Float16)proj_w[(size_t)(kt * 32 + kk) * DIMC + n];
  }
}

__global__ __launch_bounds__(256, 1)
void flash_win_kernel(const float* __restrict__ x,
                      const _Float16* __restrict__ wsW,
                      const float* __restrict__ uv_b,
                      const float* __restrict__ qq_gamma,
                      const float* __restrict__ qq_beta,
                      const float* __restrict__ qk_gamma,
                      const float* __restrict__ qk_beta,
                      const float* __restrict__ proj_b,
                      float* __restrict__ out) {
  __shared__ __align__(32) _Float16 sX[NTOK * DIMC];     // 64 KB window tile (f16)
  __shared__ __align__(32) _Float16 sQQ[NTOK * SDIM];    // 32 KB qq  -> later v^T
  __shared__ __align__(32) _Float16 sQK[NTOK * SDIM];    // 32 KB qk  -> later gate
  __shared__ __align__(32) _Float16 sAttn[NTOK * SDIM];  // 32 KB attn (f16)
  __shared__ __align__(32) _Float16 sWb[2][DIMC * 32];   // 2 x 16 KB staged weights
  __shared__ unsigned sTokBase[NTOK];

  _Float16* sVT = sQQ;
  _Float16* sG  = sQK;
  const _Float16* wsUV = wsW;
  const _Float16* wsPJ = wsW + WS_PJ_BASE;

  const int tid  = threadIdx.x;
  const int lane = tid & 31;
  const int wave = tid >> 5;
  const int mb   = wave * 16;
  const int rhi  = (lane >> 4) * 8;
  const int ln15 = lane & 15;

  // ---- window decode: wid = b*256 + ((tw*4)+hw)*8 + www --------------------
  const int wid = blockIdx.x;
  const int b   = wid >> 8;
  const int wi  = wid & 255;
  const int tw  = wi >> 5, hw = (wi >> 3) & 3, ww = wi & 7;
  if (tid < NTOK) {
    int t = tid >> 6, h = (tid >> 3) & 7, w = tid & 7;
    int tg = tw * 2 + t, hg = hw * 8 + h, wg = ww * 8 + w;
    sTokBase[tid] = (((unsigned)(b * 16 + tg) * 32 + hg) * 64 + wg) * DIMC;
  }
  __syncthreads();

  // ---- prime first weight tile (async) while loading x -> sX (f16) --------
  stage_cp<512>(sWb[0], wsUV + 1024 * 32, tid);   // base cols of K-tile 0
  for (int e = tid; e < NTOK * (DIMC / 4); e += 256) {
    int n  = e >> 6;
    int c4 = (e & 63) << 2;
    const float4 v = *(const float4*)(x + sTokBase[n] + c4);
    _Float16* d = sX + n * DIMC + c4;
    d[0] = (_Float16)v.x; d[1] = (_Float16)v.y;
    d[2] = (_Float16)v.z; d[3] = (_Float16)v.w;
  }
  stage_fence();
  __syncthreads();

  // ---- base = xw @ uv_w[:,1024:1152] (double-buffered K pipeline) ---------
  v8f acc[8] = {};
#pragma unroll
  for (int i = 0; i < 8; ++i) {
    if (i < 7) stage_cp<512>(sWb[(i + 1) & 1], wsUV + (i + 1) * WS_UV_TILE + 1024 * 32, tid);
    v16h a = ldA16(sX, DIMC, mb, i * 32, lane);
    mm_tiles<8>(acc, a, sWb[i & 1], 32, 0, lane);
    stage_fence();
    __syncthreads();
  }

  // ---- ScaleOffset -> qq / qk ---------------------------------------------
#pragma unroll
  for (int j = 0; j < 8; ++j) {
    int col = j * 16 + ln15;
    float bb = uv_b[2 * HIDD + col];
    float gq = qq_gamma[col], bq = qq_beta[col];
    float gk = qk_gamma[col], bk = qk_beta[col];
#pragma unroll
    for (int g = 0; g < 8; ++g) {
      int m = mb + g + rhi;
      float v = acc[j][g] + bb;
      sQQ[m * SDIM + col] = (_Float16)(v * gq + bq);
      sQK[m * SDIM + col] = (_Float16)(v * gk + bk);
    }
  }
  __syncthreads();

  // ---- 3D rope over (t,h,w); parts d = [44,42,42]; qq also * SCALE --------
  {
    const int offs[3] = {0, 44, 86};
    const int dh[3]   = {22, 21, 21};
    for (int e = tid; e < NTOK * 64; e += 256) {
      int n = e >> 6, i = e & 63;
      int p, ii;
      if (i < 22)      { p = 0; ii = i; }
      else if (i < 43) { p = 1; ii = i - 22; }
      else             { p = 2; ii = i - 43; }
      int pos = (p == 0) ? (n >> 6) : (p == 1) ? ((n >> 3) & 7) : (n & 7);
      float d   = 2.0f * (float)dh[p];
      float inv = __powf(10000.0f, -(2.0f * (float)ii) / d);
      float ang = (float)pos * inv;
      float cs = __cosf(ang), sn = __sinf(ang);
      int s1 = n * SDIM + offs[p] + ii;
      int s2 = s1 + dh[p];
      float a1 = (float)sQQ[s1], a2 = (float)sQQ[s2];
      sQQ[s1] = (_Float16)((a1 * cs - a2 * sn) * SCALE_Q);
      sQQ[s2] = (_Float16)((a1 * sn + a2 * cs) * SCALE_Q);
      a1 = (float)sQK[s1]; a2 = (float)sQK[s2];
      sQK[s1] = (_Float16)(a1 * cs - a2 * sn);
      sQK[s2] = (_Float16)(a1 * sn + a2 * cs);
    }
  }
  __syncthreads();

  // ---- attn = qq @ qk^T ; sq-relu / N ; store f16 -------------------------
  {
    v8f at[8] = {};
#pragma unroll
    for (int i = 0; i < 4; ++i) {
      v16h a = ldA16(sQQ, SDIM, mb, i * 32, lane);
      mm_tiles<8>(at, a, sQK, SDIM, i * 32, lane);
    }
#pragma unroll
    for (int j = 0; j < 8; ++j) {
      int col = j * 16 + ln15;
#pragma unroll
      for (int g = 0; g < 8; ++g) {
        int m = mb + g + rhi;
        float v = at[j][g];
        v = v > 0.0f ? v : 0.0f;
        sAttn[m * SDIM + col] = (_Float16)(v * v * (1.0f / 128.0f));
      }
    }
  }
  __syncthreads();   // qq/qk now dead -> buffers become sVT / sG

  // ---- hidden loop: 4 chunks of 128 over HID ------------------------------
  v8f outf[16] = {};
  for (int c = 0; c < 4; ++c) {
    // fused u+v chunk GEMM: u cols -> sW rows 0..127, v cols -> rows 128..255
    v8f uvf[16] = {};
    stage_cp<512>(sWb[0],             wsUV + c * 4096, tid);
    stage_cp<512>(sWb[0] + SDIM * 32, wsUV + 16384 + c * 4096, tid);
    stage_fence();
    __syncthreads();
#pragma unroll
    for (int i = 0; i < 8; ++i) {
      if (i < 7) {
        const _Float16* tile = wsUV + (i + 1) * WS_UV_TILE;
        stage_cp<512>(sWb[(i + 1) & 1],             tile + c * 4096, tid);
        stage_cp<512>(sWb[(i + 1) & 1] + SDIM * 32, tile + 16384 + c * 4096, tid);
      }
      v16h a = ldA16(sX, DIMC, mb, i * 32, lane);
      mm_tiles<16>(uvf, a, sWb[i & 1], 32, 0, lane);
      stage_fence();
      __syncthreads();
    }
    // u (+bias) -> sG row-major; v (+bias) -> sVT transposed (vectorized 16B)
#pragma unroll
    for (int j = 0; j < 8; ++j) {
      int col = j * 16 + ln15;
      float bb = uv_b[c * SDIM + col];
#pragma unroll
      for (int g = 0; g < 8; ++g)
        sG[(mb + g + rhi) * SDIM + col] = (_Float16)(uvf[j][g] + bb);
    }
#pragma unroll
    for (int j = 8; j < 16; ++j) {
      int col = (j - 8) * 16 + ln15;
      float bb = uv_b[HIDD + c * SDIM + col];
      v8h pk;
#pragma unroll
      for (int g = 0; g < 8; ++g) pk[g] = (_Float16)(uvf[j][g] + bb);
      *(v8h*)(sVT + col * NTOK + mb + rhi) = pk;
    }
    __syncthreads();

    // quad chunk = attn @ v_chunk ; gate with u in-place in sG
    {
      v8f qf[8] = {};
#pragma unroll
      for (int i = 0; i < 4; ++i) {
        v16h a = ldA16(sAttn, SDIM, mb, i * 32, lane);
        mm_tiles<8>(qf, a, sVT, NTOK, i * 32, lane);
      }
#pragma unroll
      for (int j = 0; j < 8; ++j) {
        int col = j * 16 + ln15;
#pragma unroll
        for (int g = 0; g < 8; ++g) {
          int idx = (mb + g + rhi) * SDIM + col;
          sG[idx] = (_Float16)((float)sG[idx] * qf[j][g]);
        }
      }
    }
    __syncthreads();

    // out += gate_chunk @ proj_w[c*128:(c+1)*128, :] (double-buffered K)
    stage_cp<1024>(sWb[0], wsPJ + (c * 4) * WS_PJ_TILE, tid);
    stage_fence();
    __syncthreads();
#pragma unroll
    for (int i = 0; i < 4; ++i) {
      if (i < 3) stage_cp<1024>(sWb[(i + 1) & 1], wsPJ + (c * 4 + i + 1) * WS_PJ_TILE, tid);
      v16h a = ldA16(sG, SDIM, mb, i * 32, lane);
      mm_tiles<16>(outf, a, sWb[i & 1], 32, 0, lane);
      stage_fence();
      __syncthreads();
    }
  }

  // ---- epilogue: +proj_b, window_reverse scatter --------------------------
#pragma unroll
  for (int j = 0; j < 16; ++j) {
    int col = j * 16 + ln15;
    float pb = proj_b[col];
#pragma unroll
    for (int g = 0; g < 8; ++g) {
      int m = mb + g + rhi;
      out[sTokBase[m] + col] = outf[j][g] + pb;
    }
  }
}

extern "C" void kernel_launch(void* const* d_in, const int* in_sizes, int n_in,
                              void* d_out, int out_size, void* d_ws, size_t ws_size,
                              hipStream_t stream) {
  (void)in_sizes; (void)n_in; (void)out_size; (void)ws_size;
  const float* x        = (const float*)d_in[0];
  const float* uv_w     = (const float*)d_in[1];
  const float* uv_b     = (const float*)d_in[2];
  const float* qq_gamma = (const float*)d_in[3];
  const float* qq_beta  = (const float*)d_in[4];
  const float* qk_gamma = (const float*)d_in[5];
  const float* qk_beta  = (const float*)d_in[6];
  const float* proj_w   = (const float*)d_in[7];
  const float* proj_b   = (const float*)d_in[8];
  float* outp = (float*)d_out;
  _Float16* ws = (_Float16*)d_ws;   // needs WS_TOTAL*2 = ~852 KB of scratch

  // 1) one-time (per launch) weight convert/transpose/tile into workspace
  prep_weights_kernel<<<(WS_TOTAL + 255) / 256, 256, 0, stream>>>(uv_w, proj_w, ws);
  // 2) 512 windows (B=2 x nW=256), one workgroup (8 wave32) per window
  flash_win_kernel<<<512, 256, 0, stream>>>(x, ws, uv_b, qq_gamma, qq_beta,
                                            qk_gamma, qk_beta, proj_b, outp);
}
